// Self_Attn_Img_83717502533832
// MI455X (gfx1250) — matrix operand order repaired and checked by reading, not compiled, not run on hardware
//
#include <hip/hip_runtime.h>
#include <hip/hip_bf16.h>

typedef __attribute__((ext_vector_type(16))) _Float16 v16h;
typedef __attribute__((ext_vector_type(8)))  float    v8f;

#define HH   55
#define PAD  64
#define WAVES 4

union HalfPack { _Float16 h[2]; unsigned int u; };
union AFrag    { v16h v; unsigned int u[8]; };

// Kernel 1: reduce the rank-1 collapse scalars:
// ws[0]=A=Wq.Wk  ws[1]=C=bq.Wk  ws[2]=E=Wv.Wo  ws[3]=F=bv.Wo+bo
__global__ void attn_scalars_kernel(const float* __restrict__ Wq, const float* __restrict__ bq,
                                    const float* __restrict__ Wk, const float* __restrict__ bk,
                                    const float* __restrict__ Wv, const float* __restrict__ bv,
                                    const float* __restrict__ Wo, const float* __restrict__ bo,
                                    float* __restrict__ ws) {
    (void)bk; // B=Wq.bk and D=bq.bk are row-constant in the softmax -> cancel
    __shared__ float red[4][64];
    int t = threadIdx.x;
    float a = 0.f, c = 0.f, e = 0.f, f = 0.f;
    if (t < HH) {
        a = Wq[t] * Wk[t];
        c = bq[t] * Wk[t];
        e = Wv[t] * Wo[t];
        f = bv[t] * Wo[t];
    }
    red[0][t] = a; red[1][t] = c; red[2][t] = e; red[3][t] = f;
    __syncthreads();
    for (int s = 32; s > 0; s >>= 1) {
        if (t < s) {
            red[0][t] += red[0][t + s];
            red[1][t] += red[1][t + s];
            red[2][t] += red[2][t + s];
            red[3][t] += red[3][t + s];
        }
        __syncthreads();
    }
    if (t == 0) {
        ws[0] = red[0][0];
        ws[1] = red[1][0];
        ws[2] = red[2][0];
        ws[3] = red[3][0] + bo[0];
    }
}

// Kernel 2: per-sample fused softmax-attention using the rank-1 collapse.
// One wave32 per sample; P.x matvec done via v_wmma_f32_16x16x32_f16.
__global__ __launch_bounds__(WAVES * 32)
void attn_main_kernel(const float* __restrict__ q,
                      const float* __restrict__ scal,
                      const float* __restrict__ gammap,
                      float* __restrict__ out, int N) {
    __shared__ _Float16 P[WAVES][PAD][PAD];   // exp-weight matrix, f16, zero-padded
    __shared__ float    xs[WAVES][PAD];       // x vector, zero-padded
    __shared__ float    rs[WAVES][PAD];       // softmax row sums (unnormalized)
    __shared__ float    wsum[WAVES][PAD];     // WMMA matvec result

    const int lane = threadIdx.x & 31;
    const int wv   = threadIdx.x >> 5;
    const int n    = blockIdx.x * WAVES + wv;
    const bool active = (n < N);

    const float A     = scal[0];
    const float C     = scal[1];
    const float E     = scal[2];
    const float Fb    = scal[3];
    const float gamma = gammap[0];

    // ---- load x (zero pad 55..63) ----
    for (int idx = lane; idx < PAD; idx += 32) {
        float v = 0.f;
        if (active && idx < HH) v = q[n * HH + idx];
        xs[wv][idx] = v;
    }
    __syncthreads();

    // ---- softmax weights: row i handled by lane (i = lane, lane+32) ----
    for (int rr = 0; rr < 2; ++rr) {
        const int i = lane + 32 * rr;
        const float ti = A * xs[wv][i] + C;
        float m = -3.4e38f;
        for (int j = 0; j < HH; ++j) m = fmaxf(m, ti * xs[wv][j]);
        float r = 0.f;
        for (int j = 0; j < HH; ++j) {
            float e = __expf(ti * xs[wv][j] - m);
            r += e;
            P[wv][i][j] = (_Float16)e;
        }
        for (int j = HH; j < PAD; ++j) P[wv][i][j] = (_Float16)0.f;
        rs[wv][i] = r;
    }
    __syncthreads();

    // ---- s = P . x via WMMA (x replicated across the 16 B-columns) ----
    const int g  = lane >> 4;   // half-wave group
    const int ln = lane & 15;

    AFrag bf[2];
    for (int ks = 0; ks < 2; ++ks) {
        const int k0 = ks * 32 + g * 16;    // B: lanes 0-15 hold K=0..15, lanes 16-31 K=16..31
        for (int v = 0; v < 8; ++v) {
            HalfPack hp;
            hp.h[0] = (_Float16)xs[wv][k0 + 2 * v];
            hp.h[1] = (_Float16)xs[wv][k0 + 2 * v + 1];
            bf[ks].u[v] = hp.u;
        }
    }

    for (int tm = 0; tm < 4; ++tm) {
        const int M = tm * 16 + ln;         // A: lane holds row M = L%16
        v8f c = {};
        for (int ks = 0; ks < 2; ++ks) {
            AFrag af;
            const int kb = ks * 32 + g * 16;
            for (int v = 0; v < 8; ++v) {
                // A k order used here (contiguous, as B): VGPR v holds K = 2*v + 16*(L/16), pairs
                const int K = kb + 2 * v;
                af.u[v] = *(const unsigned int*)&P[wv][M][K];
            }
            c = __builtin_amdgcn_wmma_f32_16x16x32_f16(
                    /*neg_a=*/false, af.v, /*neg_b=*/false, bf[ks].v,
                    /*c_mod=*/(short)0, c, /*reuse_a=*/false, /*reuse_b=*/false);
        }
        // D layout: VGPR r -> row tm*16 + r + 8*g, value replicated across N; lane ln==0 writes
        if (ln == 0) {
            for (int r = 0; r < 8; ++r) wsum[wv][tm * 16 + r + 8 * g] = c[r];
        }
    }
    __syncthreads();

    // ---- normalize + output projection + residual ----
    for (int idx = lane; idx < HH; idx += 32) {
        if (active) {
            const float s = wsum[wv][idx] / rs[wv][idx];
            out[n * HH + idx] = gamma * (E * s + Fb) + xs[wv][idx];
        }
    }
}

extern "C" void kernel_launch(void* const* d_in, const int* in_sizes, int n_in,
                              void* d_out, int out_size, void* d_ws, size_t ws_size,
                              hipStream_t stream) {
    const float* q     = (const float*)d_in[0];
    const float* Wq    = (const float*)d_in[1];
    const float* bq    = (const float*)d_in[2];
    const float* Wk    = (const float*)d_in[3];
    const float* bk    = (const float*)d_in[4];
    const float* Wv    = (const float*)d_in[5];
    const float* bv    = (const float*)d_in[6];
    const float* Wo    = (const float*)d_in[7];
    const float* bo    = (const float*)d_in[8];
    const float* gamma = (const float*)d_in[9];
    float* out = (float*)d_out;
    float* ws  = (float*)d_ws;

    const int N = in_sizes[0] / HH;   // 32768 samples of length-55 vectors

    attn_scalars_kernel<<<1, 64, 0, stream>>>(Wq, bq, Wk, bk, Wv, bv, Wo, bo, ws);

    const int blocks = (N + WAVES - 1) / WAVES;
    attn_main_kernel<<<blocks, WAVES * 32, 0, stream>>>(q, ws, gamma, out, N);
}
